// TrellisMoEMLP_3856880632421
// MI455X (gfx1250) — matrix operand order repaired and checked by reading, not compiled, not run on hardware
//
#include <hip/hip_runtime.h>
#include <hip/hip_bf16.h>

// ---------------------------------------------------------------------------
// MoE MLP (top-4 of 16 experts + shared expert) for MI455X (gfx1250).
// bf16 WMMA (V_WMMA_F32_16X16X32_BF16) with fp32 accumulation; weights are
// converted fp32->bf16 on the fly while streaming (compute-bound workload,
// bf16 matrix path is ~8x the fp32 WMMA rate). Weight stream gets
// global_prefetch_b8 two K-steps ahead.
// Pipeline: router -> scan -> compact -> x:fp32->bf16 -> expert GEMM1(g,u)+SiLU
//           -> expert GEMM2 -> shared GEMM1 -> shared GEMM2 (writes out)
//           -> deterministic combine (out += sum_k w_k * y_rows[slot]).
// Workspace use: ~116 MB.
// ---------------------------------------------------------------------------

#define H_DIM  2048
#define E_NUM  16
#define F_DIM  1408
#define FS_DIM 4096
#define T_TOK  2048   // B*S
#define TOPK   4
#define NSLOTS (T_TOK * TOPK)

typedef __attribute__((ext_vector_type(16))) __bf16 v16bf;
typedef __attribute__((ext_vector_type(8)))  float  v8f;

// ---- WMMA fragment helpers -------------------------------------------------

// A fragment, 16x32 bf16, row-major source (k order matches B fragment):
// lanes 0-15 : row M = lane   , K = 0..15
// lanes 16-31: row M = lane-16, K = 16..31
__device__ __forceinline__ v16bf load_a_frag(const __bf16* __restrict__ row,
                                             int k0, int half) {
  union { v16bf v; uint4 q[2]; } a;
  a.q[0] = *(const uint4*)(row + k0 + half * 16);
  a.q[1] = *(const uint4*)(row + k0 + half * 16 + 8);
  return a.v;
}

// B fragment, 32x16 bf16, built from fp32 row-major W (ld = row stride).
// lanes 0-15 : col N = lane   , K = 0..15
// lanes 16-31: col N = lane-16, K = 16..31
// Prefetches the same fragment 2 K-steps ahead (global_prefetch_b8 -> GL2).
__device__ __forceinline__ v16bf load_b_frag_f32(const float* __restrict__ W,
                                                 int ld, int k0, int half,
                                                 int col) {
  v16bf b = {};
  const float* p = W + (size_t)(k0 + half * 16) * ld + col;
  __builtin_prefetch(p + (size_t)64 * ld, 0, 1);   // next-next K tile
#pragma unroll
  for (int i = 0; i < 16; ++i) {
    b[i] = (__bf16)p[(size_t)i * ld];   // v_cvt_pk_bf16_f32 on the fly
  }
  return b;
}

__device__ __forceinline__ v8f wmma_bf16(v16bf a, v16bf b, v8f c) {
  return __builtin_amdgcn_wmma_f32_16x16x32_bf16(false, a, false, b, (short)0,
                                                 c, false, false);
}

// Fast SiLU: z * rcp(1 + exp(-z)) using v_rcp_f32 (avoids the IEEE div chain).
__device__ __forceinline__ float silu_f(float z) {
  return z * __builtin_amdgcn_rcpf(1.0f + __expf(-z));
}

// ---- small setup kernels ---------------------------------------------------

__global__ void init_kernel(int* counts, int* cursor) {
  int i = threadIdx.x;
  if (i < E_NUM) { counts[i] = 0; cursor[i] = 0; }
}

// One wave per token: logits, softmax, top-4, normalized weights.
__global__ __launch_bounds__(256) void router_kernel(
    const float* __restrict__ x, const float* __restrict__ wr,
    int* __restrict__ topk_idx, float* __restrict__ topk_w,
    int* __restrict__ counts) {
  const int wid = threadIdx.x >> 5, lane = threadIdx.x & 31;
  const int t = blockIdx.x * 8 + wid;
  const float* xr = x + (size_t)t * H_DIM;
  float acc[E_NUM];
#pragma unroll
  for (int e = 0; e < E_NUM; ++e) acc[e] = 0.0f;
  for (int h = lane; h < H_DIM; h += 32) {
    const float xv = xr[h];
#pragma unroll
    for (int e = 0; e < E_NUM; ++e) acc[e] += xv * wr[(size_t)e * H_DIM + h];
  }
#pragma unroll
  for (int e = 0; e < E_NUM; ++e) {
    float v = acc[e];
#pragma unroll
    for (int off = 16; off >= 1; off >>= 1) v += __shfl_xor(v, off, 32);
    acc[e] = v;
  }
  // softmax (redundant per lane; trivial)
  float mx = acc[0];
#pragma unroll
  for (int e = 1; e < E_NUM; ++e) mx = fmaxf(mx, acc[e]);
  float pr[E_NUM], se = 0.0f;
#pragma unroll
  for (int e = 0; e < E_NUM; ++e) { pr[e] = __expf(acc[e] - mx); se += pr[e]; }
  const float inv = 1.0f / se;
#pragma unroll
  for (int e = 0; e < E_NUM; ++e) pr[e] *= inv;
  // top-4 (ties -> lowest index, matches lax.top_k)
  int ids[TOPK]; float wsv[TOPK]; float wsum = 0.0f;
#pragma unroll
  for (int k = 0; k < TOPK; ++k) {
    int bj = 0; float bv = -1.0f;
#pragma unroll
    for (int e = 0; e < E_NUM; ++e) if (pr[e] > bv) { bv = pr[e]; bj = e; }
    ids[k] = bj; wsv[k] = bv; pr[bj] = -2.0f; wsum += bv;
  }
  const float wn = 1.0f / wsum;
  if (lane == 0) {
#pragma unroll
    for (int k = 0; k < TOPK; ++k) {
      topk_idx[t * TOPK + k] = ids[k];
      topk_w[t * TOPK + k]   = wsv[k] * wn;
      atomicAdd(&counts[ids[k]], 1);
    }
  }
}

__global__ void scan_kernel(const int* __restrict__ counts,
                            int* __restrict__ offsets) {
  if (threadIdx.x == 0) {
    int s = 0;
    for (int e = 0; e < E_NUM; ++e) { offsets[e] = s; s += counts[e]; }
    offsets[E_NUM] = s;
  }
}

__global__ void compact_kernel(const int* __restrict__ topk_idx,
                               const float* __restrict__ topk_w,
                               const int* __restrict__ offsets,
                               int* __restrict__ cursor,
                               int* __restrict__ expert_rows,
                               float* __restrict__ row_w,
                               int* __restrict__ slot_idx) {
  const int t = blockIdx.x * 256 + threadIdx.x;
  if (t >= T_TOK) return;
#pragma unroll
  for (int k = 0; k < TOPK; ++k) {
    const int e = topk_idx[t * TOPK + k];
    const int pos = atomicAdd(&cursor[e], 1);
    const int slot = offsets[e] + pos;
    expert_rows[slot] = t;
    row_w[slot] = topk_w[t * TOPK + k];
    slot_idx[t * TOPK + k] = slot;
  }
}

__global__ void convert_x_kernel(const float* __restrict__ x,
                                 __bf16* __restrict__ xb) {
  const size_t i = (size_t)blockIdx.x * 256 + threadIdx.x;
  xb[i] = (__bf16)x[i];
}

// ---- expert GEMM stage 1: g = x@Wg, u = x@Wu, h = silu(w*g)*(w*u) ----------
// grid (F/64, T/32, E), block 256 = 8 waves as 2(M) x 4(N).
__global__ __launch_bounds__(256) void moe_stage1_kernel(
    const __bf16* __restrict__ xb, const float* __restrict__ Wg_all,
    const float* __restrict__ Wu_all, const int* __restrict__ offsets,
    const int* __restrict__ expert_rows, const float* __restrict__ row_w,
    __bf16* __restrict__ h_rows) {
  const int e = blockIdx.z;
  const int base = offsets[e];
  const int cnt = offsets[e + 1] - base;
  const int m0b = blockIdx.y * 32;
  if (m0b >= cnt) return;                      // block-uniform early exit
  const int wid = threadIdx.x >> 5, lane = threadIdx.x & 31;
  const int half = lane >> 4, l16 = lane & 15;
  const int m0 = m0b + (wid >> 2) * 16;
  const int n0 = blockIdx.x * 64 + (wid & 3) * 16;
  const int col = n0 + l16;
  // gather: A-row of this lane = list position m0 + (lane&15), clamped
  const int p = m0 + l16;
  const int pc = (p < cnt) ? p : (cnt - 1);
  const int tok = expert_rows[base + pc];
  const __bf16* arow = xb + (size_t)tok * H_DIM;
  const float* Wg = Wg_all + (size_t)e * H_DIM * F_DIM;
  const float* Wu = Wu_all + (size_t)e * H_DIM * F_DIM;

  v8f cg = {}, cu = {};
  for (int k0 = 0; k0 < H_DIM; k0 += 32) {
    const v16bf a  = load_a_frag(arow, k0, half);
    const v16bf bg = load_b_frag_f32(Wg, F_DIM, k0, half, col);
    const v16bf bu = load_b_frag_f32(Wu, F_DIM, k0, half, col);
    cg = wmma_bf16(a, bg, cg);
    cu = wmma_bf16(a, bu, cu);
  }
  // epilogue: D row = v + 8*half, col = n0 + (lane&15)
#pragma unroll
  for (int v = 0; v < 8; ++v) {
    const int prow = m0 + v + 8 * half;
    if (prow < cnt) {
      const int slot = base + prow;
      const float w = row_w[slot];
      const float g = w * cg[v], u = w * cu[v];
      h_rows[(size_t)slot * F_DIM + col] = (__bf16)(silu_f(g) * u);
    }
  }
}

// ---- expert GEMM stage 2: y = h @ Wd  (per-slot, weight applied later) -----
// grid (H/64, T/32, E)
__global__ __launch_bounds__(256) void moe_stage2_kernel(
    const __bf16* __restrict__ h_rows, const float* __restrict__ Wd_all,
    const int* __restrict__ offsets, float* __restrict__ y_rows) {
  const int e = blockIdx.z;
  const int base = offsets[e];
  const int cnt = offsets[e + 1] - base;
  const int m0b = blockIdx.y * 32;
  if (m0b >= cnt) return;
  const int wid = threadIdx.x >> 5, lane = threadIdx.x & 31;
  const int half = lane >> 4, l16 = lane & 15;
  const int m0 = m0b + (wid >> 2) * 16;
  const int n0 = blockIdx.x * 64 + (wid & 3) * 16;
  const int col = n0 + l16;
  const int p = m0 + l16;
  const int pc = (p < cnt) ? p : (cnt - 1);
  const __bf16* arow = h_rows + (size_t)(base + pc) * F_DIM;
  const float* Wd = Wd_all + (size_t)e * F_DIM * H_DIM;

  v8f c = {};
  for (int k0 = 0; k0 < F_DIM; k0 += 32) {
    const v16bf a = load_a_frag(arow, k0, half);
    const v16bf b = load_b_frag_f32(Wd, H_DIM, k0, half, col);
    c = wmma_bf16(a, b, c);
  }
#pragma unroll
  for (int v = 0; v < 8; ++v) {
    const int prow = m0 + v + 8 * half;
    if (prow < cnt) y_rows[(size_t)(base + prow) * H_DIM + col] = c[v];
  }
}

// ---- shared expert stage 1: hs = silu(x@Wg_s) * (x@Wu_s) -------------------
// grid (FS/64, T/32)
__global__ __launch_bounds__(256) void shared_stage1_kernel(
    const __bf16* __restrict__ xb, const float* __restrict__ Wg_s,
    const float* __restrict__ Wu_s, __bf16* __restrict__ hs) {
  const int wid = threadIdx.x >> 5, lane = threadIdx.x & 31;
  const int half = lane >> 4, l16 = lane & 15;
  const int m0 = blockIdx.y * 32 + (wid >> 2) * 16;
  const int n0 = blockIdx.x * 64 + (wid & 3) * 16;
  const int col = n0 + l16;
  const __bf16* arow = xb + (size_t)(m0 + l16) * H_DIM;
  v8f cg = {}, cu = {};
  for (int k0 = 0; k0 < H_DIM; k0 += 32) {
    const v16bf a  = load_a_frag(arow, k0, half);
    const v16bf bg = load_b_frag_f32(Wg_s, FS_DIM, k0, half, col);
    const v16bf bu = load_b_frag_f32(Wu_s, FS_DIM, k0, half, col);
    cg = wmma_bf16(a, bg, cg);
    cu = wmma_bf16(a, bu, cu);
  }
#pragma unroll
  for (int v = 0; v < 8; ++v) {
    const int t = m0 + v + 8 * half;
    hs[(size_t)t * FS_DIM + col] = (__bf16)(silu_f(cg[v]) * cu[v]);
  }
}

// ---- shared expert stage 2: out = hs @ Wd_s (overwrites d_out) -------------
// grid (H/64, T/32)
__global__ __launch_bounds__(256) void shared_stage2_kernel(
    const __bf16* __restrict__ hs, const float* __restrict__ Wd_s,
    float* __restrict__ out) {
  const int wid = threadIdx.x >> 5, lane = threadIdx.x & 31;
  const int half = lane >> 4, l16 = lane & 15;
  const int m0 = blockIdx.y * 32 + (wid >> 2) * 16;
  const int n0 = blockIdx.x * 64 + (wid & 3) * 16;
  const int col = n0 + l16;
  const __bf16* arow = hs + (size_t)(m0 + l16) * FS_DIM;
  v8f c = {};
  for (int k0 = 0; k0 < FS_DIM; k0 += 32) {
    const v16bf a = load_a_frag(arow, k0, half);
    const v16bf b = load_b_frag_f32(Wd_s, H_DIM, k0, half, col);
    c = wmma_bf16(a, b, c);
  }
#pragma unroll
  for (int v = 0; v < 8; ++v) {
    const int t = m0 + v + 8 * half;
    out[(size_t)t * H_DIM + col] = c[v];
  }
}

// ---- deterministic combine: out += sum_k w_k * y_rows[slot_k] --------------
__global__ void combine_kernel(const int* __restrict__ slot_idx,
                               const float* __restrict__ topk_w,
                               const float* __restrict__ y_rows,
                               float* __restrict__ out) {
  const size_t idx = (size_t)blockIdx.x * 256 + threadIdx.x;
  const int t = (int)(idx / H_DIM);
  const int h = (int)(idx % H_DIM);
  float s = out[idx];
#pragma unroll
  for (int k = 0; k < TOPK; ++k) {
    const int slot = slot_idx[t * TOPK + k];
    const float w = topk_w[t * TOPK + k];
    s += w * y_rows[(size_t)slot * H_DIM + h];
  }
  out[idx] = s;
}

// ---------------------------------------------------------------------------

extern "C" void kernel_launch(void* const* d_in, const int* in_sizes, int n_in,
                              void* d_out, int out_size, void* d_ws,
                              size_t ws_size, hipStream_t stream) {
  const float* x   = (const float*)d_in[0];
  const float* wr  = (const float*)d_in[1];
  const float* Wg  = (const float*)d_in[2];
  const float* Wu  = (const float*)d_in[3];
  const float* Wd  = (const float*)d_in[4];
  const float* Wgs = (const float*)d_in[5];
  const float* Wus = (const float*)d_in[6];
  const float* Wds = (const float*)d_in[7];
  float* out = (float*)d_out;
  (void)in_sizes; (void)n_in; (void)out_size; (void)ws_size;

  char* ws = (char*)d_ws;
  size_t off = 0;
  auto wsalloc = [&](size_t bytes) -> char* {
    char* p = ws + off;
    off = (off + bytes + 255) & ~(size_t)255;
    return p;
  };
  int*    counts      = (int*)wsalloc(E_NUM * sizeof(int));
  int*    cursor      = (int*)wsalloc(E_NUM * sizeof(int));
  int*    offsets     = (int*)wsalloc((E_NUM + 1) * sizeof(int));
  int*    topk_idx    = (int*)wsalloc((size_t)T_TOK * TOPK * sizeof(int));
  float*  topk_w      = (float*)wsalloc((size_t)T_TOK * TOPK * sizeof(float));
  int*    slot_idx    = (int*)wsalloc((size_t)T_TOK * TOPK * sizeof(int));
  int*    expert_rows = (int*)wsalloc((size_t)NSLOTS * sizeof(int));
  float*  row_w       = (float*)wsalloc((size_t)NSLOTS * sizeof(float));
  __bf16* xb          = (__bf16*)wsalloc((size_t)T_TOK * H_DIM * sizeof(__bf16));
  __bf16* h_rows      = (__bf16*)wsalloc((size_t)NSLOTS * F_DIM * sizeof(__bf16));
  float*  y_rows      = (float*)wsalloc((size_t)NSLOTS * H_DIM * sizeof(float));
  __bf16* hs          = (__bf16*)wsalloc((size_t)T_TOK * FS_DIM * sizeof(__bf16));

  init_kernel<<<1, 64, 0, stream>>>(counts, cursor);
  router_kernel<<<T_TOK / 8, 256, 0, stream>>>(x, wr, topk_idx, topk_w, counts);
  scan_kernel<<<1, 32, 0, stream>>>(counts, offsets);
  compact_kernel<<<T_TOK / 256, 256, 0, stream>>>(topk_idx, topk_w, offsets,
                                                  cursor, expert_rows, row_w,
                                                  slot_idx);
  convert_x_kernel<<<(T_TOK * H_DIM) / 256, 256, 0, stream>>>(x, xb);

  moe_stage1_kernel<<<dim3(F_DIM / 64, T_TOK / 32, E_NUM), 256, 0, stream>>>(
      xb, Wg, Wu, offsets, expert_rows, row_w, h_rows);
  moe_stage2_kernel<<<dim3(H_DIM / 64, T_TOK / 32, E_NUM), 256, 0, stream>>>(
      h_rows, Wd, offsets, y_rows);

  shared_stage1_kernel<<<dim3(FS_DIM / 64, T_TOK / 32), 256, 0, stream>>>(
      xb, Wgs, Wus, hs);
  shared_stage2_kernel<<<dim3(H_DIM / 64, T_TOK / 32), 256, 0, stream>>>(
      hs, Wds, out);

  combine_kernel<<<(T_TOK * H_DIM) / 256, 256, 0, stream>>>(slot_idx, topk_w,
                                                            y_rows, out);
}
